// SimplifiedGAT_6803228197705
// MI455X (gfx1250) — matrix operand order repaired and checked by reading, not compiled, not run on hardware
//
#include <hip/hip_runtime.h>

// ---------------------------------------------------------------------------
// SimplifiedGAT on MI455X (gfx1250, wave32, WMMA).
//
// Math restructure: scores depend only on column j =>
//   v[j] = exp(e[j] - max(e));  num = adj @ (v*h);  den = adj @ v;
//   out[i,:] = num[i,:] / den[i]
// Dominant cost: adj (268 MB int32, read exactly once, ~11.5 us @ 23.3 TB/s)
// feeding an 8192x8192x144 f16 WMMA GEMM (17.2 GFLOP + 12.5% for the fused
// denominator tile).  B operand is pre-swizzled into the exact WMMA lane
// layout and staged to LDS with GLOBAL_LOAD_ASYNC_TO_LDS when available.
// ---------------------------------------------------------------------------

typedef __attribute__((ext_vector_type(16))) _Float16 v16h;
typedef __attribute__((ext_vector_type(8)))  _Float16 v8h;
typedef __attribute__((ext_vector_type(8)))  float    v8f;
typedef __attribute__((ext_vector_type(4)))  int      v4i;

// address-space qualified vector pointers for the async global->LDS builtin
typedef v4i __attribute__((address_space(1)))* gv4p;   // global
typedef v4i __attribute__((address_space(3)))* lv4p;   // LDS

#define N_NODES 8192
#define FIN     256
#define FOUT    128
#define FOUTP   144      // FOUT + 16 (extra tile carrying v for the denominator)
#define NTILES  9        // FOUTP / 16
#define LEAK    0.2f
#define KSPLIT  8

// workspace byte offsets (total ~6.5 MB)
#define OFF_H   0u         // h      : 8192*128 f32  (4 MB)
#define OFF_E   4194304u   // e      : 8192 f32
#define OFF_M   4227072u   // max(e) : 1 f32
#define OFF_DEN 4227328u   // den    : 8192 f32
#define OFF_B   4260096u   // Bsw    : swizzled f16 B matrix (2.25 MB)

// ---- async global->LDS path (ASYNCcnt), guarded so compilation never breaks
#if defined(__has_builtin)
#if __has_builtin(__builtin_amdgcn_global_load_async_to_lds_b128) && \
    __has_builtin(__builtin_amdgcn_s_wait_asynccnt)
#define USE_ASYNC_LDS 1
#endif
#endif
#ifndef USE_ASYNC_LDS
#define USE_ASYNC_LDS 0
#endif

__device__ __forceinline__ v8f wmma_f16(v16h a, v16h b, v8f c) {
    return __builtin_amdgcn_wmma_f32_16x16x32_f16(
        /*neg_a=*/false, a, /*neg_b=*/false, b,
        /*c_mod=*/(short)0, c, /*reuse_a=*/false, /*reuse_b=*/false);
}

// ---------------------------------------------------------------------------
// Kernel 1: h = X @ W   (f16 in / f32 accumulate).  One wave per 16x16 tile.
// A frag (16x32 f16, ISA 7.12.2): lane L holds row M=L&15;
//   kbase=(L>>4)*8; halves 0..7 -> K=kbase+i; halves 8..15 -> K=kbase+16+(i-8)
// B frag (32x16): lanes 0-15 col=lane, K=0..7,16..23; lanes 16-31 col=lane-16, K=8..15,24..31
// ---------------------------------------------------------------------------
__global__ __launch_bounds__(256) void k_gemm_xw(const float* __restrict__ x,
                                                 const float* __restrict__ W,
                                                 float* __restrict__ h) {
    const int lane    = threadIdx.x & 31;
    const int wave    = threadIdx.x >> 5;
    const int rowTile = blockIdx.x;   // 512 row tiles
    const int colTile = wave;         // 8 col tiles
    const int r16   = lane & 15;
    const int khalf = lane >> 4;      // 0 or 1
    const int row   = rowTile * 16 + r16;
    const int col   = colTile * 16 + r16;

    v8f acc = {0.f, 0.f, 0.f, 0.f, 0.f, 0.f, 0.f, 0.f};

    for (int k0 = 0; k0 < FIN; k0 += 32) {
        // ---- A fragment from x (f32 -> f16) ----
        const float4* ap = (const float4*)(x + (size_t)row * FIN + k0 + khalf * 8);
        float4 a0 = ap[0], a1 = ap[1];   // K = kb .. kb+7
        float4 a2 = ap[4], a3 = ap[5];   // K = kb+16 .. kb+23
        float af[16] = {a0.x, a0.y, a0.z, a0.w, a1.x, a1.y, a1.z, a1.w,
                        a2.x, a2.y, a2.z, a2.w, a3.x, a3.y, a3.z, a3.w};
        v16h A;
#pragma unroll
        for (int i = 0; i < 16; ++i) A[i] = (_Float16)af[i];

        // ---- B fragment from W ----
        const int kcol = k0 + khalf * 8;
        v16h B;
#pragma unroll
        for (int i = 0; i < 16; ++i)
            B[i] = (_Float16)W[(size_t)(kcol + i + (i >= 8 ? 8 : 0)) * FOUT + col];

        acc = wmma_f16(A, B, acc);
    }

    // C/D layout: VGPR r -> M = r + (lane>=16 ? 8 : 0), N = lane&15
#pragma unroll
    for (int r = 0; r < 8; ++r)
        h[(size_t)(rowTile * 16 + r + khalf * 8) * FOUT + col] = acc[r];
}

// ---------------------------------------------------------------------------
// Kernel 2: e[j] = leakyrelu(h[j,:] . a)  -- one wave per row
// ---------------------------------------------------------------------------
__global__ __launch_bounds__(256) void k_alpha_e(const float* __restrict__ h,
                                                 const float* __restrict__ a,
                                                 float* __restrict__ e) {
    const int lane = threadIdx.x & 31;
    const int row  = blockIdx.x * 8 + (threadIdx.x >> 5);
    float s = 0.f;
#pragma unroll
    for (int k = lane; k < FOUT; k += 32)
        s += h[(size_t)row * FOUT + k] * a[k];
#pragma unroll
    for (int m = 16; m; m >>= 1) s += __shfl_xor(s, m, 32);
    if (lane == 0) e[row] = (s > 0.f) ? s : LEAK * s;
}

// ---------------------------------------------------------------------------
// Kernel 3: global max of e (softmax is shift-invariant -> global max is valid)
// ---------------------------------------------------------------------------
__global__ __launch_bounds__(1024) void k_max(const float* __restrict__ e,
                                              float* __restrict__ M) {
    __shared__ float sm[1024];
    float m = -3.4e38f;
    for (int i = threadIdx.x; i < N_NODES; i += 1024) m = fmaxf(m, e[i]);
    sm[threadIdx.x] = m;
    __syncthreads();
    for (int s = 512; s; s >>= 1) {
        if (threadIdx.x < (unsigned)s)
            sm[threadIdx.x] = fmaxf(sm[threadIdx.x], sm[threadIdx.x + s]);
        __syncthreads();
    }
    if (threadIdx.x == 0) M[0] = sm[0];
}

// ---------------------------------------------------------------------------
// Kernel 4: build pre-swizzled f16 B matrix for the big GEMM.
//   B[j, c] = c<128 ? v[j]*h[j,c] : (c==128 ? v[j] : 0),  v[j]=exp(e[j]-M)
// Swizzle so each lane later loads its 16 halves contiguously:
//   idx = ((kblk*9 + ctile)*32 + lane)*16 + half
//   lane = (c&15) + (j%32 >= 16 ? 16 : 0),  half = j%16,  kblk = j/32
// ---------------------------------------------------------------------------
__global__ __launch_bounds__(256) void k_build_b(const float* __restrict__ h,
                                                 const float* __restrict__ e,
                                                 const float* __restrict__ M,
                                                 _Float16* __restrict__ Bsw) {
    int idx = blockIdx.x * 256 + threadIdx.x;
    if (idx >= N_NODES * FOUTP) return;
    int j = idx / FOUTP;
    int c = idx - j * FOUTP;
    float v   = __expf(e[j] - M[0]);
    float val = (c < FOUT) ? v * h[(size_t)j * FOUT + c]
                           : ((c == FOUT) ? v : 0.f);
    int kblk = j >> 5, kk = j & 31;
    int ctile = c >> 4, cl = c & 15;
    int lanep = cl + (kk & 16);
    int hh    = kk & 15;
    size_t sidx = (((size_t)kblk * NTILES + ctile) * 32 + lanep) * 16 + hh;
    Bsw[sidx] = (_Float16)val;
}

// ---------------------------------------------------------------------------
// Kernel 5: zero the atomic accumulators (num lives in d_out, den in ws)
// ---------------------------------------------------------------------------
__global__ __launch_bounds__(256) void k_zero(float* __restrict__ num,
                                              float* __restrict__ den) {
    int idx = blockIdx.x * 256 + threadIdx.x;
    if (idx < N_NODES * FOUT) num[idx] = 0.f;
    if (idx < N_NODES)        den[idx] = 0.f;
}

// ---------------------------------------------------------------------------
// Kernel 6: the heavy GEMM  O = f16(adj) @ Bsw   (8192 x 8192 x 144).
// Block = 256 thr (8 waves) -> 128 rows x 144 cols; adj staged coalesced
// through VGPRs with an integer 0/1 -> f16 pack trick; Bsw chunk staged via
// async global->LDS b128 (ASYNCcnt) when available.  K split 8 ways over
// blockIdx.y, accumulated with f32 atomics.
// ---------------------------------------------------------------------------
__global__ __launch_bounds__(256) void k_gemm_attn(const int* __restrict__ adj,
                                                   const _Float16* __restrict__ Bsw,
                                                   float* __restrict__ num,
                                                   float* __restrict__ den) {
    __shared__ __align__(16) _Float16 Atile[128 * 32];         // 8 KB
    __shared__ __align__(16) _Float16 Btile[NTILES * 32 * 16]; // 9 KB

    const int t    = threadIdx.x;
    const int lane = t & 31;
    const int wave = t >> 5;
    const int rowBase = blockIdx.x * 128;
    const int kblk0 = blockIdx.y * (N_NODES / 32 / KSPLIT);
    const int kblk1 = kblk0 + (N_NODES / 32 / KSPLIT);

    // WMMA A-fragment coordinates for this lane
    const int r16   = lane & 15;
    const int khalf = lane >> 4;
    const int lrow  = wave * 16 + r16;   // local row 0..127
    const int kb    = khalf * 16;

    v8f acc[NTILES];
#pragma unroll
    for (int i = 0; i < NTILES; ++i)
        acc[i] = (v8f){0.f, 0.f, 0.f, 0.f, 0.f, 0.f, 0.f, 0.f};

    // staging coordinates: 2 threads per adj row, 16 int32 each
    const int sr = t >> 1;            // 0..127
    const int sc = (t & 1) * 16;      // 0 or 16
    const size_t adjRow = (size_t)(rowBase + sr) * N_NODES;

    for (int kblk = kblk0; kblk < kblk1; ++kblk) {
        const int k0 = kblk * 32;
        __syncthreads();

        // ---- stage swizzled B chunk (9216 B, already WMMA layout) into LDS
#if USE_ASYNC_LDS
        {
            const char* src = (const char*)(Bsw + (size_t)kblk * (NTILES * 32 * 16));
            char* dst = (char*)Btile;
#pragma unroll
            for (int i = 0; i < 3; ++i) {
                int idx = i * 256 + t;            // 576 x 16B chunks
                if (idx < NTILES * 32 * 16 * 2 / 16)
                    __builtin_amdgcn_global_load_async_to_lds_b128(
                        (gv4p)(src + idx * 16), (lv4p)(dst + idx * 16), 0, 0);
            }
        }
#else
        {
            uint32_t* BtileU = (uint32_t*)Btile;
            const uint32_t* src =
                (const uint32_t*)(Bsw + (size_t)kblk * (NTILES * 32 * 16));
#pragma unroll
            for (int i = 0; i < NTILES; ++i)
                BtileU[i * 256 + t] = src[i * 256 + t];
        }
#endif

        // ---- stage adj tile: int32 0/1 -> f16 via integer pack (1.0h=0x3C00)
        {
            const v4i* gp = (const v4i*)(adj + adjRow + k0 + sc);
            v4i q0 = gp[0], q1 = gp[1], q2 = gp[2], q3 = gp[3];
            uint4 s0, s1;
            s0.x = (uint32_t)(q0.x | (q0.y << 16)) * 0x3C00u;
            s0.y = (uint32_t)(q0.z | (q0.w << 16)) * 0x3C00u;
            s0.z = (uint32_t)(q1.x | (q1.y << 16)) * 0x3C00u;
            s0.w = (uint32_t)(q1.z | (q1.w << 16)) * 0x3C00u;
            s1.x = (uint32_t)(q2.x | (q2.y << 16)) * 0x3C00u;
            s1.y = (uint32_t)(q2.z | (q2.w << 16)) * 0x3C00u;
            s1.z = (uint32_t)(q3.x | (q3.y << 16)) * 0x3C00u;
            s1.w = (uint32_t)(q3.z | (q3.w << 16)) * 0x3C00u;
            uint4* sp = (uint4*)&Atile[sr * 32 + sc];
            sp[0] = s0;
            sp[1] = s1;
            if (kblk + 1 < kblk1)  // global_prefetch_b8 for next adj chunk
                __builtin_prefetch(adj + adjRow + k0 + 32 + sc, 0, 1);
        }

#if USE_ASYNC_LDS
        __builtin_amdgcn_s_wait_asynccnt(0);
#endif
        __syncthreads();

        // ---- A fragment: two ds_load_b128 ----
        const v8h* lp = (const v8h*)&Atile[lrow * 32 + kb];
        v8h alo = lp[0];
        v8h ahi = lp[1];  // +8 halves
        v16h A = __builtin_shufflevector(alo, ahi, 0, 1, 2, 3, 4, 5, 6, 7,
                                         8, 9, 10, 11, 12, 13, 14, 15);

        // ---- 9 column tiles: B fragment + WMMA each ----
#pragma unroll
        for (int tt = 0; tt < NTILES; ++tt) {
            const v8h* bp = (const v8h*)&Btile[(tt * 32 + lane) * 16];
            v16h B = __builtin_shufflevector(bp[0], bp[1], 0, 1, 2, 3, 4, 5, 6,
                                             7, 8, 9, 10, 11, 12, 13, 14, 15);
            acc[tt] = wmma_f16(A, B, acc[tt]);
        }
    }

    // ---- accumulate partials: tiles 0..7 -> num, tile 8 col 0 -> den ----
#pragma unroll
    for (int tt = 0; tt < 8; ++tt) {
#pragma unroll
        for (int r = 0; r < 8; ++r) {
            int row = rowBase + wave * 16 + r + khalf * 8;
            atomicAdd(&num[(size_t)row * FOUT + tt * 16 + r16], acc[tt][r]);
        }
    }
    if (r16 == 0) {
#pragma unroll
        for (int r = 0; r < 8; ++r) {
            int row = rowBase + wave * 16 + r + khalf * 8;
            atomicAdd(&den[row], acc[8][r]);
        }
    }
}

// ---------------------------------------------------------------------------
// Kernel 7: out = num / den  (den==0 cannot happen for p(edge)=0.5, N=8192)
// ---------------------------------------------------------------------------
__global__ __launch_bounds__(256) void k_div(float* __restrict__ out,
                                             const float* __restrict__ den) {
    int idx = blockIdx.x * 256 + threadIdx.x;
    int row = idx >> 7;
    float d = den[row];
    out[idx] = (d != 0.f) ? out[idx] / d : 0.f;
}

// ---------------------------------------------------------------------------
extern "C" void kernel_launch(void* const* d_in, const int* in_sizes, int n_in,
                              void* d_out, int out_size, void* d_ws, size_t ws_size,
                              hipStream_t stream) {
    (void)in_sizes; (void)n_in; (void)out_size; (void)ws_size;
    const float* x   = (const float*)d_in[0];
    const int*   adj = (const int*)d_in[1];
    const float* W   = (const float*)d_in[2];
    const float* a   = (const float*)d_in[3];
    float* out = (float*)d_out;

    char* ws = (char*)d_ws;
    float*    h   = (float*)(ws + OFF_H);
    float*    e   = (float*)(ws + OFF_E);
    float*    M   = (float*)(ws + OFF_M);
    float*    den = (float*)(ws + OFF_DEN);
    _Float16* Bsw = (_Float16*)(ws + OFF_B);

    k_gemm_xw<<<512, 256, 0, stream>>>(x, W, h);
    k_alpha_e<<<1024, 256, 0, stream>>>(h, a, e);
    k_max<<<1, 1024, 0, stream>>>(e, M);
    k_build_b<<<(N_NODES * FOUTP + 255) / 256, 256, 0, stream>>>(h, e, M, Bsw);
    k_zero<<<(N_NODES * FOUT + 255) / 256, 256, 0, stream>>>(out, den);
    dim3 grid(N_NODES / 128, KSPLIT);
    k_gemm_attn<<<grid, 256, 0, stream>>>(adj, Bsw, out, den);
    k_div<<<(N_NODES * FOUT + 255) / 256, 256, 0, stream>>>(out, den);
}
